// MultiHeadLatentAttention_15513421873271
// MI455X (gfx1250) — matrix operand order repaired and checked by reading, not compiled, not run on hardware
//
#include <hip/hip_runtime.h>
#include <stdint.h>

// ---------------------------------------------------------------------------
// Types for CDNA5 WMMA (wave32, 16x16x32 bf16 -> f32)
// ---------------------------------------------------------------------------
typedef __attribute__((ext_vector_type(16))) __bf16   v16bf;
typedef __attribute__((ext_vector_type(8)))  float    v8f;
typedef __attribute__((ext_vector_type(8)))  unsigned v8u;
typedef __attribute__((ext_vector_type(4)))  unsigned u32x4;
typedef __attribute__((ext_vector_type(4)))  int      i32x4;
typedef __attribute__((ext_vector_type(8)))  int      i32x8;

#if __has_builtin(__builtin_amdgcn_tensor_load_to_lds) && __has_builtin(__builtin_amdgcn_s_wait_tensorcnt)
#define USE_TDM 1
#else
#define USE_TDM 0
#endif

__device__ __forceinline__ uint16_t f32_to_bf16(float f) {
    unsigned u = __builtin_bit_cast(unsigned, f);
    unsigned r = u + 0x7FFFu + ((u >> 16) & 1u);   // round-to-nearest-even
    return (uint16_t)(r >> 16);
}
__device__ __forceinline__ float bf16_to_f32(uint16_t h) {
    unsigned u = ((unsigned)h) << 16;
    return __builtin_bit_cast(float, u);
}

__device__ __forceinline__ v16bf make_frag(uint4 a, uint4 b) {
    v8u u;
    u[0] = a.x; u[1] = a.y; u[2] = a.z; u[3] = a.w;
    u[4] = b.x; u[5] = b.y; u[6] = b.z; u[7] = b.w;
    return __builtin_bit_cast(v16bf, u);
}
// A-fragment: two contiguous 8xbf16 chunks (lane-half selects K offset)
__device__ __forceinline__ v16bf load_frag2(const uint16_t* p0, const uint16_t* p1) {
    return make_frag(*(const uint4*)p0, *(const uint4*)p1);
}
// B-fragment: two 8xbf16 chunks at p and p+16 (lane-half selects K offset)
__device__ __forceinline__ v16bf load_frag16(const uint16_t* p) {
    return make_frag(*(const uint4*)p, *(const uint4*)(p + 16));
}
__device__ __forceinline__ v8f wmma_bf16(v16bf a, v16bf b, v8f c) {
    return __builtin_amdgcn_wmma_f32_16x16x32_bf16(false, a, false, b,
                                                   (short)0, c, false, false);
}

__device__ __forceinline__ void store_out(float* p, float v)    { *p = v; }
__device__ __forceinline__ void store_out(uint16_t* p, float v) { *p = f32_to_bf16(v); }

// ---------------------------------------------------------------------------
// TDM: stage a [128 rows x 32 cols] bf16 tile of W (row stride K elems) into
// LDS with +16B pad after every 64B (LDS pitch = 80B = 40 uint16) so that
// B-fragment ds_load_b128 reads are bank-conflict free.
// ---------------------------------------------------------------------------
#define BT_PITCH 40               // uint16 per staged row (32 data + 8 pad)

#if USE_TDM
__device__ __forceinline__ void tdm_load_b_tile(const uint16_t* gsrc,
                                                uint16_t* ldst,
                                                int rowsLeft, int K) {
    uint64_t gaddr = (uint64_t)(uintptr_t)gsrc;
    uint32_t laddr = (uint32_t)(uintptr_t)ldst;     // LDS byte offset (low 32)
    u32x4 g0;
    g0[0] = 1u;                                        // count=1
    g0[1] = laddr;                                     // lds_addr
    g0[2] = (uint32_t)(gaddr & 0xFFFFFFFFu);           // global_addr[31:0]
    g0[3] = (uint32_t)((gaddr >> 32) & 0x01FFFFFFu)    // global_addr[56:32]
          | (2u << 30);                                // type = 2 (image)
    i32x8 g1;
    // data_size=1(2B) | pad_enable | pad_interval=3 (16 dw = 64B) | pad_amount=3 (4 dw = 16B)
    g1[0] = (int)((1u << 16) | (1u << 20) | (3u << 22) | (3u << 25));
    g1[1] = (int)(((uint32_t)K & 0xFFFFu) << 16);                      // tensor_dim0[15:0]
    g1[2] = (int)((((uint32_t)K >> 16) & 0xFFFFu)                      // tensor_dim0[31:16]
                | (((uint32_t)rowsLeft & 0xFFFFu) << 16));             // tensor_dim1[15:0]
    g1[3] = (int)(((((uint32_t)rowsLeft) >> 16) & 0xFFFFu)             // tensor_dim1[31:16]
                | (32u << 16));                                        // tile_dim0 = 32
    g1[4] = (int)128u;                                                 // tile_dim1 = 128
    g1[5] = (int)K;                                                    // tensor_dim0_stride[31:0]
    g1[6] = 0;                                                         // stride hi / dim1_stride
    g1[7] = 0;
    i32x4 z4 = {0, 0, 0, 0};
    i32x8 z8 = {0, 0, 0, 0, 0, 0, 0, 0};
    // 6-arg form (clang-23 / therock-10.0 headers): g0, g1, g2, g3, g4, cpol
    __builtin_amdgcn_tensor_load_to_lds(g0, g1, z4, z4, z8, 0);
}
#endif

__device__ __forceinline__ void stage_b_tile(uint16_t* dst, const uint16_t* src,
                                             int rowsLeft, int K, int tid) {
#if USE_TDM
    if ((tid >> 5) == 0) tdm_load_b_tile(src, dst, rowsLeft, K);
    (void)tid;
#else
    const int row = tid >> 1, half = tid & 1;          // 256 thr: 128 rows x 2 halves
    const uint16_t* s = src + (size_t)row * K + half * 16;
    uint16_t* d = dst + row * BT_PITCH + half * 16;
    *(uint4*)d       = *(const uint4*)s;
    *(uint4*)(d + 8) = *(const uint4*)(s + 8);
    (void)rowsLeft;
#endif
}

__device__ __forceinline__ void stage_wait(int tid) {
#if USE_TDM
    if ((tid >> 5) == 0) __builtin_amdgcn_s_wait_tensorcnt(0);
#else
    (void)tid;
#endif
}

// ---------------------------------------------------------------------------
// f32 -> bf16 conversion (grid-stride)
// ---------------------------------------------------------------------------
__global__ void k_f32_to_bf16(const float* __restrict__ in,
                              uint16_t* __restrict__ out, size_t n) {
    size_t i = (size_t)blockIdx.x * blockDim.x + threadIdx.x;
    size_t stride = (size_t)gridDim.x * blockDim.x;
    for (; i < n; i += stride) out[i] = f32_to_bf16(in[i]);
}

// ---------------------------------------------------------------------------
// Big GEMM: C[M,N] = A[M,K] @ W[N,K]^T. Block 256 thr = 8 waves (4m x 2n),
// wave tile 32x64 (8 WMMAs / K-step), block tile 128x128, K-step 32.
// W tile double-buffered in LDS via TDM (DMA overlaps WMMA issue).
// Requires M%128==0, N%128==0, K%32==0.
// ---------------------------------------------------------------------------
template <typename OutT>
__global__ __launch_bounds__(256)
void k_gemm_big(const uint16_t* __restrict__ A, const uint16_t* __restrict__ W,
                OutT* __restrict__ C, int M, int N, int K) {
    __shared__ uint16_t sB[2][128 * BT_PITCH];

    const int tid  = threadIdx.x;
    const int lane = tid & 31;
    const int wave = tid >> 5;
    const int wm   = wave >> 1;            // 0..3 : 32-row sub-tile
    const int wn   = wave & 1;             // 0..1 : 64-col sub-tile
    const int ln   = lane & 15;
    const int lh   = lane >> 4;

    const int bm = blockIdx.y * 128;
    const int bn = blockIdx.x * 128;

    const uint16_t* Arow0 = A + (size_t)(bm + wm * 32 + ln) * K + lh * 8;
    const uint16_t* Arow1 = Arow0 + (size_t)16 * K;
    const uint16_t* Wblk  = W + (size_t)bn * K;
    const int rowsLeft    = N - bn;

    v8f acc[2][4];
    const v8f vz = {};
#pragma unroll
    for (int i = 0; i < 2; i++)
#pragma unroll
        for (int j = 0; j < 4; j++) acc[i][j] = vz;

    const int nk = K / 32;
    stage_b_tile(sB[0], Wblk, rowsLeft, K, tid);           // prologue: buf 0

    for (int kt = 0; kt < nk; kt++) {
        const int cur = kt & 1;
        stage_wait(tid);               // wave0: its DMA into `cur` is complete
        __syncthreads();               // `cur` visible to (and unread by) all
        if (kt + 1 < nk)
            stage_b_tile(sB[cur ^ 1], Wblk + (size_t)(kt + 1) * 32,
                         rowsLeft, K, tid);

        const int k = kt * 32;
        __builtin_prefetch(Arow0 + k + 128, 0, 1);         // global_prefetch_b8
        v16bf a0 = load_frag2(Arow0 + k, Arow0 + k + 16);
        v16bf a1 = load_frag2(Arow1 + k, Arow1 + k + 16);
#pragma unroll
        for (int nf = 0; nf < 4; nf++) {
            const uint16_t* bp =
                &sB[cur][(wn * 64 + nf * 16 + ln) * BT_PITCH + lh * 8];
            v16bf bfrag = load_frag16(bp);
            acc[0][nf] = wmma_bf16(a0, bfrag, acc[0][nf]);
            acc[1][nf] = wmma_bf16(a1, bfrag, acc[1][nf]);
        }
    }

#pragma unroll
    for (int mf = 0; mf < 2; mf++)
#pragma unroll
        for (int nf = 0; nf < 4; nf++)
#pragma unroll
            for (int r = 0; r < 8; r++) {
                int m = bm + wm * 32 + mf * 16 + r + 8 * lh;
                int n = bn + wn * 64 + nf * 16 + ln;
                store_out(C + (size_t)m * N + n, acc[mf][nf][r]);
            }
}

// ---------------------------------------------------------------------------
// Small GEMM (N multiple of 64 only; used for the N=64 k_rope projection).
// ---------------------------------------------------------------------------
template <typename OutT>
__global__ __launch_bounds__(256)
void k_gemm_small(const uint16_t* __restrict__ A, const uint16_t* __restrict__ W,
                  OutT* __restrict__ C, int M, int N, int K) {
    const int lane = threadIdx.x & 31;
    const int wave = threadIdx.x >> 5;
    const int wm   = wave >> 1;
    const int wn   = wave & 1;
    const int ln   = lane & 15;
    const int lh   = lane >> 4;

    const int bm = blockIdx.y * 64;
    const int bn = blockIdx.x * 64;

    const int row = bm + wm * 16 + ln;
    const int n0  = bn + wn * 32 + ln;

    const uint16_t* Arow  = A + (size_t)row * K + lh * 8;
    const uint16_t* Wrow0 = W + (size_t)n0 * K + lh * 8;
    const uint16_t* Wrow1 = W + (size_t)(n0 + 16) * K + lh * 8;

    v8f c0 = {}, c1 = {};
#pragma unroll 4
    for (int k = 0; k < K; k += 32) {
        v16bf a  = load_frag2(Arow + k, Arow + k + 16);
        v16bf b0 = load_frag16(Wrow0 + k);
        v16bf b1 = load_frag16(Wrow1 + k);
        c0 = wmma_bf16(a, b0, c0);
        c1 = wmma_bf16(a, b1, c1);
    }
#pragma unroll
    for (int r = 0; r < 8; r++) {
        int m = bm + wm * 16 + r + 8 * lh;
        store_out(C + (size_t)m * N + n0,      c0[r]);
        store_out(C + (size_t)m * N + n0 + 16, c1[r]);
    }
}

// ---------------------------------------------------------------------------
// RMSNorm over last dim R, f32 in -> bf16 out. One block per row.
// ---------------------------------------------------------------------------
__global__ __launch_bounds__(256)
void k_rmsnorm_bf16(const float* __restrict__ X, const float* __restrict__ g,
                    uint16_t* __restrict__ Y, int R) {
    const int row = blockIdx.x;
    const float* x = X + (size_t)row * R;
    __shared__ float red[256];
    float s = 0.f;
    for (int c = threadIdx.x; c < R; c += 256) { float v = x[c]; s += v * v; }
    red[threadIdx.x] = s;
    __syncthreads();
    for (int off = 128; off > 0; off >>= 1) {
        if ((int)threadIdx.x < off) red[threadIdx.x] += red[threadIdx.x + off];
        __syncthreads();
    }
    const float inv = rsqrtf(red[0] / (float)R + 1e-6f);
    for (int c = threadIdx.x; c < R; c += 256)
        Y[(size_t)row * R + c] = f32_to_bf16(x[c] * inv * g[c]);
}

// ---------------------------------------------------------------------------
// Assemble Q [B,H,S,128]: nope from q_nope GEMM; rope half = rotate(k_rope)
// (broadcast over heads — matches reference's q_rot/k_rot swap).
// ---------------------------------------------------------------------------
__global__ void k_assemble_q(const uint16_t* __restrict__ qn,   // [tok,1024]
                             const uint16_t* __restrict__ kr,   // [tok,64]
                             const float* __restrict__ fcos,    // [S,32]
                             const float* __restrict__ fsin,
                             uint16_t* __restrict__ Q) {
    size_t idx = (size_t)blockIdx.x * 256 + threadIdx.x;  // [b][h][s][d]
    int d = idx & 127;
    size_t t = idx >> 7;
    int s = (int)(t & 2047); t >>= 11;
    int h = (int)(t & 15);
    int b = (int)(t >> 4);
    size_t tok = (size_t)b * 2048 + s;
    uint16_t val;
    if (d < 64) {
        val = qn[tok * 1024 + h * 64 + d];
    } else {
        int j = d - 64, p = j >> 1;
        float xe = bf16_to_f32(kr[tok * 64 + 2 * p]);
        float xo = bf16_to_f32(kr[tok * 64 + 2 * p + 1]);
        float c = fcos[s * 32 + p], sn = fsin[s * 32 + p];
        float o = (j & 1) ? (xe * sn + xo * c) : (xe * c - xo * sn);
        val = f32_to_bf16(o);
    }
    Q[idx] = val;
}

// Assemble K [B,H,S,128]: nope from k_nope GEMM; rope half = rotate(q_rope[h]).
__global__ void k_assemble_k(const uint16_t* __restrict__ kn,   // [tok,1024]
                             const uint16_t* __restrict__ qr,   // [tok,1024]
                             const float* __restrict__ fcos,
                             const float* __restrict__ fsin,
                             uint16_t* __restrict__ K) {
    size_t idx = (size_t)blockIdx.x * 256 + threadIdx.x;
    int d = idx & 127;
    size_t t = idx >> 7;
    int s = (int)(t & 2047); t >>= 11;
    int h = (int)(t & 15);
    int b = (int)(t >> 4);
    size_t tok = (size_t)b * 2048 + s;
    uint16_t val;
    if (d < 64) {
        val = kn[tok * 1024 + h * 64 + d];
    } else {
        int j = d - 64, p = j >> 1;
        float xe = bf16_to_f32(qr[tok * 1024 + h * 64 + 2 * p]);
        float xo = bf16_to_f32(qr[tok * 1024 + h * 64 + 2 * p + 1]);
        float c = fcos[s * 32 + p], sn = fsin[s * 32 + p];
        float o = (j & 1) ? (xe * sn + xo * c) : (xe * c - xo * sn);
        val = f32_to_bf16(o);
    }
    K[idx] = val;
}

// Vt [B,H,128,S] = transpose of v GEMM [tok, h*128+d] (for PV WMMA B-frags).
__global__ void k_assemble_vt(const uint16_t* __restrict__ vb,
                              uint16_t* __restrict__ Vt) {
    size_t idx = (size_t)blockIdx.x * 256 + threadIdx.x;  // [b][h][d][s]
    int s = (int)(idx & 2047);
    size_t t = idx >> 11;
    int d = (int)(t & 127); t >>= 7;
    int h = (int)(t & 15);
    int b = (int)(t >> 4);
    Vt[idx] = vb[((size_t)b * 2048 + s) * 2048 + h * 128 + d];
}

// ---------------------------------------------------------------------------
// Flash attention. 1 wave per (b, h, 16-query tile). Keys in tiles of 32.
// ---------------------------------------------------------------------------
__global__ __launch_bounds__(32)
void k_attention(const uint16_t* __restrict__ Q, const uint16_t* __restrict__ Kb,
                 const uint16_t* __restrict__ Vt, uint16_t* __restrict__ Ob,
                 float scale) {
    __shared__ float    sS[16][32];
    __shared__ uint16_t sP[16][32];
    __shared__ float    sAlpha[16];
    __shared__ float    sInvL[16];

    const int lane = threadIdx.x;
    const int ln = lane & 15, lh = lane >> 4;
    const int qt = blockIdx.x, h = blockIdx.y, b = blockIdx.z;

    const size_t headOfs = (((size_t)b * 16 + h) * 2048) * 128;
    const uint16_t* Qh = Q  + headOfs;   // [S,128]
    const uint16_t* Kh = Kb + headOfs;   // [S,128]
    const uint16_t* Vh = Vt + headOfs;   // [128,S]

    const int qrow = qt * 16 + ln;
    v16bf qf[4];
#pragma unroll
    for (int dc = 0; dc < 4; dc++) {
        const uint16_t* p = Qh + (size_t)qrow * 128 + dc * 32 + lh * 8;
        qf[dc] = load_frag2(p, p + 16);
    }

    v8f o[8];
    const v8f vzero = {};
#pragma unroll
    for (int i = 0; i < 8; i++) o[i] = vzero;
    float m_i = -1e30f, l_i = 0.f;

    const int ktmax = (qt * 16 + 15) / 32;
    for (int kt = 0; kt <= ktmax; kt++) {
        const int kBase = kt * 32;
        __builtin_prefetch(Kh + (size_t)(kBase + 32 + ln) * 128, 0, 1);

        // ---- scores: 16q x 32k
        v8f sc[2]; sc[0] = vzero; sc[1] = vzero;
#pragma unroll
        for (int nt = 0; nt < 2; nt++) {
            const int krow = kBase + nt * 16 + ln;
            const uint16_t* Kp = Kh + (size_t)krow * 128 + lh * 8;
#pragma unroll
            for (int dc = 0; dc < 4; dc++)
                sc[nt] = wmma_bf16(qf[dc], load_frag16(Kp + dc * 32), sc[nt]);
        }
        // scale + causal mask -> LDS (C-layout: lane=col, reg=row)
#pragma unroll
        for (int nt = 0; nt < 2; nt++)
#pragma unroll
            for (int r = 0; r < 8; r++) {
                int m = r + 8 * lh;
                int kg = kBase + nt * 16 + ln;
                float s = sc[nt][r] * scale;
                if (kg > qt * 16 + m) s = -1e30f;
                sS[m][nt * 16 + ln] = s;
            }
        __syncthreads();

        // ---- streaming softmax: lane i owns query row i
        if (lane < 16) {
            const int m = lane;
            float mx = m_i;
#pragma unroll
            for (int j = 0; j < 32; j++) mx = fmaxf(mx, sS[m][j]);
            float alpha = __expf(m_i - mx);
            float sum = 0.f;
#pragma unroll
            for (int j = 0; j < 32; j++) {
                float e = __expf(sS[m][j] - mx);
                sum += e;
                sP[m][j] = f32_to_bf16(e);
            }
            l_i = l_i * alpha + sum;
            m_i = mx;
            sAlpha[m] = alpha;
        }
        __syncthreads();

        // rescale running output
        float al[8];
#pragma unroll
        for (int r = 0; r < 8; r++) al[r] = sAlpha[r + 8 * lh];
#pragma unroll
        for (int f = 0; f < 8; f++)
#pragma unroll
            for (int r = 0; r < 8; r++) o[f][r] = o[f][r] * al[r];

        // ---- P @ V: reload P as A-fragment, Vt rows give contiguous B-frags
        const uint16_t* pp = &sP[ln][lh * 8];
        v16bf pf = load_frag2(pp, pp + 16);
#pragma unroll
        for (int dc = 0; dc < 8; dc++) {
            const uint16_t* Vp = Vh + (size_t)(dc * 16 + ln) * 2048 + kBase + lh * 8;
            o[dc] = wmma_bf16(pf, load_frag16(Vp), o[dc]);
        }
        __syncthreads();
    }

    if (lane < 16) sInvL[lane] = 1.0f / l_i;
    __syncthreads();
    float il[8];
#pragma unroll
    for (int r = 0; r < 8; r++) il[r] = sInvL[r + 8 * lh];

#pragma unroll
    for (int dc = 0; dc < 8; dc++)
#pragma unroll
        for (int r = 0; r < 8; r++) {
            int m = r + 8 * lh;
            int qg = qt * 16 + m;
            Ob[((size_t)(b * 2048 + qg)) * 2048 + h * 128 + dc * 16 + ln] =
                f32_to_bf16(o[dc][r] * il[r]);
        }
}

// ---------------------------------------------------------------------------
// Host side
// ---------------------------------------------------------------------------
extern "C" void kernel_launch(void* const* d_in, const int* in_sizes, int n_in,
                              void* d_out, int out_size, void* d_ws, size_t ws_size,
                              hipStream_t stream) {
    (void)in_sizes; (void)n_in; (void)out_size; (void)ws_size;
    constexpr int B = 2, S = 2048, D = 2048, H = 16, HD = 128;
    constexpr int QR = 1536, KVR = 512, ND = 64, RD = 64;
    constexpr int TOK = B * S;                       // 4096

    const float* x      = (const float*)d_in[0];
    const float* fcos   = (const float*)d_in[2];
    const float* fsin   = (const float*)d_in[3];
    const float* w_cq   = (const float*)d_in[4];
    const float* w_dqn  = (const float*)d_in[5];
    const float* w_dqr  = (const float*)d_in[6];
    const float* w_ckv  = (const float*)d_in[7];
    const float* w_dkn  = (const float*)d_in[8];
    const float* w_dv   = (const float*)d_in[9];
    const float* w_kr   = (const float*)d_in[10];
    const float* g_q    = (const float*)d_in[11];
    const float* g_kv   = (const float*)d_in[12];
    const float* w_proj = (const float*)d_in[13];

    char* ws = (char*)d_ws;
    size_t off = 0;
    auto alloc = [&](size_t bytes) -> void* {
        void* p = ws + off;
        off = (off + bytes + 255) & ~(size_t)255;
        return p;
    };

    uint16_t* xb      = (uint16_t*)alloc((size_t)TOK * D * 2);
    uint16_t* wb_cq   = (uint16_t*)alloc((size_t)QR * D * 2);
    uint16_t* wb_dqn  = (uint16_t*)alloc((size_t)ND * H * QR * 2);
    uint16_t* wb_dqr  = (uint16_t*)alloc((size_t)RD * H * QR * 2);
    uint16_t* wb_ckv  = (uint16_t*)alloc((size_t)KVR * D * 2);
    uint16_t* wb_dkn  = (uint16_t*)alloc((size_t)ND * H * KVR * 2);
    uint16_t* wb_dv   = (uint16_t*)alloc((size_t)HD * H * KVR * 2);
    uint16_t* wb_kr   = (uint16_t*)alloc((size_t)RD * D * 2);
    uint16_t* wb_proj = (uint16_t*)alloc((size_t)D * (H * HD) * 2);
    float*    cq_f    = (float*)   alloc((size_t)TOK * QR * 4);
    uint16_t* cq_b    = (uint16_t*)alloc((size_t)TOK * QR * 2);
    float*    ckv_f   = (float*)   alloc((size_t)TOK * KVR * 4);
    uint16_t* ckv_b   = (uint16_t*)alloc((size_t)TOK * KVR * 2);
    uint16_t* qn_b    = (uint16_t*)alloc((size_t)TOK * ND * H * 2);
    uint16_t* qr_b    = (uint16_t*)alloc((size_t)TOK * RD * H * 2);
    uint16_t* kn_b    = (uint16_t*)alloc((size_t)TOK * ND * H * 2);
    uint16_t* v_b     = (uint16_t*)alloc((size_t)TOK * HD * H * 2);
    uint16_t* kr_b    = (uint16_t*)alloc((size_t)TOK * RD * 2);
    uint16_t* Qb      = (uint16_t*)alloc((size_t)B * H * S * HD * 2);
    uint16_t* Kb      = (uint16_t*)alloc((size_t)B * H * S * HD * 2);
    uint16_t* Vtb     = (uint16_t*)alloc((size_t)B * H * HD * S * 2);
    uint16_t* att_b   = (uint16_t*)alloc((size_t)TOK * (H * HD) * 2);

    auto cvt = [&](const float* src, uint16_t* dst, size_t n) {
        k_f32_to_bf16<<<2048, 256, 0, stream>>>(src, dst, n);
    };
    cvt(x,      xb,      (size_t)TOK * D);
    cvt(w_cq,   wb_cq,   (size_t)QR * D);
    cvt(w_dqn,  wb_dqn,  (size_t)ND * H * QR);
    cvt(w_dqr,  wb_dqr,  (size_t)RD * H * QR);
    cvt(w_ckv,  wb_ckv,  (size_t)KVR * D);
    cvt(w_dkn,  wb_dkn,  (size_t)ND * H * KVR);
    cvt(w_dv,   wb_dv,   (size_t)HD * H * KVR);
    cvt(w_kr,   wb_kr,   (size_t)RD * D);
    cvt(w_proj, wb_proj, (size_t)D * H * HD);

    dim3 blk(256);
    auto bgrid = [](int M, int N) { return dim3(N / 128, M / 128); };

    // cq = rmsnorm(x @ w_cq^T)
    k_gemm_big<float><<<bgrid(TOK, QR), blk, 0, stream>>>(xb, wb_cq, cq_f, TOK, QR, D);
    k_rmsnorm_bf16<<<TOK, blk, 0, stream>>>(cq_f, g_q, cq_b, QR);
    // q_nope / q_rope
    k_gemm_big<uint16_t><<<bgrid(TOK, ND * H), blk, 0, stream>>>(cq_b, wb_dqn, qn_b, TOK, ND * H, QR);
    k_gemm_big<uint16_t><<<bgrid(TOK, RD * H), blk, 0, stream>>>(cq_b, wb_dqr, qr_b, TOK, RD * H, QR);
    // ckv = rmsnorm(x @ w_ckv^T)
    k_gemm_big<float><<<bgrid(TOK, KVR), blk, 0, stream>>>(xb, wb_ckv, ckv_f, TOK, KVR, D);
    k_rmsnorm_bf16<<<TOK, blk, 0, stream>>>(ckv_f, g_kv, ckv_b, KVR);
    // k_nope / v / k_rope
    k_gemm_big<uint16_t><<<bgrid(TOK, ND * H), blk, 0, stream>>>(ckv_b, wb_dkn, kn_b, TOK, ND * H, KVR);
    k_gemm_big<uint16_t><<<bgrid(TOK, HD * H), blk, 0, stream>>>(ckv_b, wb_dv, v_b, TOK, HD * H, KVR);
    k_gemm_small<uint16_t><<<dim3(RD / 64, TOK / 64), blk, 0, stream>>>(xb, wb_kr, kr_b, TOK, RD, D);

    // RoPE + head-major assembly
    const size_t qkElems = (size_t)B * H * S * HD;           // 8.4M
    k_assemble_q<<<(unsigned)(qkElems / 256), blk, 0, stream>>>(qn_b, kr_b, fcos, fsin, Qb);
    k_assemble_k<<<(unsigned)(qkElems / 256), blk, 0, stream>>>(kn_b, qr_b, fcos, fsin, Kb);
    k_assemble_vt<<<(unsigned)(qkElems / 256), blk, 0, stream>>>(v_b, Vtb);

    // flash attention: one wave per (b, h, 16-query tile)
    k_attention<<<dim3(S / 16, H, B), dim3(32), 0, stream>>>(
        Qb, Kb, Vtb, att_b, 1.0f / sqrtf((float)HD));

    // output projection -> f32 d_out
    k_gemm_big<float><<<bgrid(TOK, D), blk, 0, stream>>>(
        att_b, wb_proj, (float*)d_out, TOK, D, H * HD);
}
